// ConvFrameDecoder_11063835754769
// MI455X (gfx1250) — matrix operand order repaired and checked by reading, not compiled, not run on hardware
//
#include <hip/hip_runtime.h>
#include <hip/hip_bf16.h>
#include <math.h>

// ---------------------------------------------------------------------------
// MI455X / gfx1250 implementation of ConvFrameDecoder.
//  * Visual encoder hoisted out of the recurrence: 3 big fp16 WMMA GEMMs.
//  * enc converted to fp16 once (134 MB -> resident in 192 MB L2).
//  * Wave32 WMMA GEMM, templated on wave N-fragments:
//      NF=4: block tile 128x128, wave 32x64, 8 v_wmma per K-step
//      NF=2: block tile 128x64,  wave 32x32 (for N=64 conv2)
//    Double-buffered LDS: global->reg loads for tile k+1 overlap the WMMAs of
//    tile k; one s_barrier per K-step. Fused bias/BN/ReLU epilogue.
// ---------------------------------------------------------------------------

typedef __attribute__((ext_vector_type(16))) _Float16 v16h;
typedef __attribute__((ext_vector_type(8)))  _Float16 v8h;
typedef __attribute__((ext_vector_type(8)))  float    v8f;

union Frag16 { v16h v; v8h h[2]; };

// ------------------------------- small utility kernels ---------------------

__global__ void k_cvt_h(const float* __restrict__ s, _Float16* __restrict__ d, int n) {
  int i = blockIdx.x * 256 + threadIdx.x;
  if (i < n) d[i] = (_Float16)s[i];
}

__global__ void k_wcat(const float* __restrict__ wih, const float* __restrict__ whh,
                       _Float16* __restrict__ d, int N, int K1, int K2) {
  int i = blockIdx.x * 256 + threadIdx.x;
  int K = K1 + K2;
  if (i < N * K) {
    int n = i / K, k = i % K;
    float v = (k < K1) ? wih[(size_t)n * K1 + k] : whh[(size_t)n * K2 + (k - K1)];
    d[i] = (_Float16)v;
  }
}

// fold eval-mode BN (+ conv bias) into per-channel scale/shift:
// y = relu( (x@W^T) * s + t ),  s = g*rsqrt(v+eps),  t = beta - m*s + bias*s
__global__ void k_bnprep(const float* __restrict__ g, const float* __restrict__ bb,
                         const float* __restrict__ m, const float* __restrict__ v,
                         const float* __restrict__ bias,
                         float* __restrict__ s, float* __restrict__ t, int n) {
  int i = blockIdx.x * 256 + threadIdx.x;
  if (i < n) {
    float sc = g[i] * rsqrtf(v[i] + 1e-5f);
    s[i] = sc;
    t[i] = bb[i] - m[i] * sc + bias[i] * sc;
  }
}

__global__ void k_addb(const float* __restrict__ a, const float* __restrict__ b,
                       float* __restrict__ d, int n) {
  int i = blockIdx.x * 256 + threadIdx.x;
  if (i < n) d[i] = a[i] + b[i];
}

__global__ void k_init(const float* __restrict__ h0, const float* __restrict__ c0,
                       const float* __restrict__ go,
                       _Float16* __restrict__ xcat, float* __restrict__ hF,
                       float* __restrict__ cF, int B) {
  int i = blockIdx.x * 256 + threadIdx.x;
  if (i < B * 1024) {
    int b = i >> 10, d = i & 1023;
    hF[i] = h0[i];
    cF[i] = c0[i];
    xcat[(size_t)b * 2688 + 1664 + d] = (_Float16)h0[i];   // h slot
  }
  if (i < B * 128) {
    int b = i >> 7, j = i & 127;
    xcat[(size_t)b * 2688 + 1536 + j] = (_Float16)go[j];   // e0 = GO token
  }
}

__global__ void k_copy(const float* __restrict__ s, float* __restrict__ d, int n) {
  int i = blockIdx.x * 256 + threadIdx.x;
  if (i < n) d[i] = s[i];
}

// ------------------------------- WMMA GEMM ---------------------------------

// global -> registers, A side.
// amode 0: A fp16 row-major (aPtr pre-offset to row, +akc). amode 1: gather
// fp32 frames [bt,512,49] as A(row = bt*49+hw, col = c) via frBase (K-invariant).
__device__ __forceinline__ void ld_a(const _Float16* aPtr, const float* frBase,
                                     int amode, int kt, int akc, v8h (&r)[2]) {
  if (amode == 0) {
    r[0] = *(const v8h*)(aPtr + kt);
    r[1] = *(const v8h*)(aPtr + kt + 8);
  } else {
#pragma unroll
    for (int j = 0; j < 8; ++j) r[0][j] = (_Float16)frBase[(size_t)(kt + akc + j) * 49];
#pragma unroll
    for (int j = 0; j < 8; ++j) r[1][j] = (_Float16)frBase[(size_t)(kt + akc + 8 + j) * 49];
  }
}

// C[M,N] = epilogue( A[M,K] (fp16, lda) @ W[N,K]^T (fp16) )
// NF = N-fragments per wave (2 -> block 128x64, 4 -> block 128x128).
// omode (outH): 0 = row-major [M,N]; 1 = conv-flatten: [bt, n*49+hw]
template<int NF>
__global__ __launch_bounds__(256) void k_gemm(
    const _Float16* __restrict__ A, int lda,
    const float* __restrict__ AF,
    const _Float16* __restrict__ W,
    const float* __restrict__ scale, const float* __restrict__ shift,
    float* __restrict__ outF, int ldF,
    _Float16* __restrict__ outH,
    int M, int N, int K, int relu, int amode, int omode)
{
  constexpr int BN = 32 * NF;               // block tile N (64 or 128)
  __shared__ _Float16 As[2][128 * 40];      // 128 rows x 32 K, stride 40
  __shared__ _Float16 Bs[2][BN * 40];

  const int tid  = threadIdx.x;
  const int lane = tid & 31;
  const int wave = tid >> 5;
  const int wr   = wave & 3;                // wave row (M), 4 waves
  const int wc   = wave >> 2;               // wave col (N), 2 waves
  const int mBase = blockIdx.y * 128;
  const int nBase = blockIdx.x * BN;

  v8f acc[2][NF] = {};

  // A staging: 128x32 halfs, 16 per thread
  const int arow = tid >> 1;                // 0..127
  const int akc  = (tid & 1) * 16;          // 0 or 16
  // B staging: BN x 32 halfs
  const int brow = (NF == 4) ? (tid >> 1) : (tid >> 2);
  const int bkc  = (NF == 4) ? (tid & 1) * 16 : (tid & 3) * 8;

  const _Float16* aPtr = A ? (A + (size_t)(mBase + arow) * lda + akc) : nullptr;
  const _Float16* bPtr = W + (size_t)(nBase + brow) * K + bkc;
  const float* frBase = nullptr;
  if (amode == 1) {
    int m = mBase + arow;
    frBase = AF + (size_t)(m / 49) * (512 * 49) + (m % 49);
  }

  v8h aR[2], bR[2];

  // prologue: stage tile 0
  ld_a(aPtr, frBase, amode, 0, akc, aR);
  bR[0] = *(const v8h*)(bPtr);
  if (NF == 4) bR[1] = *(const v8h*)(bPtr + 8);
  *(v8h*)&As[0][arow * 40 + akc]     = aR[0];
  *(v8h*)&As[0][arow * 40 + akc + 8] = aR[1];
  *(v8h*)&Bs[0][brow * 40 + bkc]     = bR[0];
  if (NF == 4) *(v8h*)&Bs[0][brow * 40 + bkc + 8] = bR[1];
  __syncthreads();

  int cur = 0;
  for (int kt = 0; kt < K; kt += 32) {
    const bool hasNext = (kt + 32 < K);
    // issue global loads for tile k+1 (latency hides under the WMMAs below)
    if (hasNext) {
      ld_a(aPtr, frBase, amode, kt + 32, akc, aR);
      bR[0] = *(const v8h*)(bPtr + kt + 32);
      if (NF == 4) bR[1] = *(const v8h*)(bPtr + kt + 40);
    }

    // ---- fragments per the 16-bit WMMA VGPR layout:
    // A: lanes 0-15 hold K 0..7 & 16..23, lanes 16-31 hold K 8..15 & 24..31
    // B: lane n%16 = column, lanes 0-15 hold K 0..7 & 16..23, lanes 16-31 hold K 8..15 & 24..31
    Frag16 af[2], bf[NF];
#pragma unroll
    for (int mi = 0; mi < 2; ++mi) {
      int mrow = wr * 32 + mi * 16 + (lane & 15);
      const _Float16* pa = &As[cur][mrow * 40 + ((lane >> 4) << 3)];
      af[mi].h[0] = *(const v8h*)pa;
      af[mi].h[1] = *(const v8h*)(pa + 16);
    }
#pragma unroll
    for (int ni = 0; ni < NF; ++ni) {
      int nrow = wc * (16 * NF) + ni * 16 + (lane & 15);
      const _Float16* pb = &Bs[cur][nrow * 40 + ((lane >> 4) << 3)];
      bf[ni].h[0] = *(const v8h*)pb;
      bf[ni].h[1] = *(const v8h*)(pb + 16);
    }
#pragma unroll
    for (int mi = 0; mi < 2; ++mi)
#pragma unroll
      for (int ni = 0; ni < NF; ++ni)
        acc[mi][ni] = __builtin_amdgcn_wmma_f32_16x16x32_f16(
            false, af[mi].v, false, bf[ni].v, (short)0, acc[mi][ni], false, false);

    if (hasNext) {
      // store tile k+1 into the other buffer; one barrier per K-step
      int nxt = cur ^ 1;
      *(v8h*)&As[nxt][arow * 40 + akc]     = aR[0];
      *(v8h*)&As[nxt][arow * 40 + akc + 8] = aR[1];
      *(v8h*)&Bs[nxt][brow * 40 + bkc]     = bR[0];
      if (NF == 4) *(v8h*)&Bs[nxt][brow * 40 + bkc + 8] = bR[1];
      __syncthreads();
      cur = nxt;
    }
  }

  // ---- epilogue (C layout: VGPR v -> M = v + 8*(lane/16), N = lane%16)
#pragma unroll
  for (int mi = 0; mi < 2; ++mi)
#pragma unroll
    for (int ni = 0; ni < NF; ++ni) {
      int n = nBase + wc * (16 * NF) + ni * 16 + (lane & 15);
      float sc = scale ? scale[n] : 1.0f;
      float sh = shift ? shift[n] : 0.0f;
#pragma unroll
      for (int v = 0; v < 8; ++v) {
        int m = mBase + wr * 32 + mi * 16 + ((lane >> 4) << 3) + v;
        float x = acc[mi][ni][v] * sc + sh;
        if (relu) x = fmaxf(x, 0.0f);
        if (outF) outF[(size_t)m * ldF + n] = x;
        if (outH) {
          if (omode == 0) outH[(size_t)m * N + n] = (_Float16)x;
          else            outH[(size_t)(m / 49) * 3136 + n * 49 + (m % 49)] = (_Float16)x;
        }
      }
    }
}

// ------------------------------- attention ---------------------------------

// raw[b,l] = enc[b,l,:] . q[b,:]; softmax over L; write scores (f32 output).
__global__ __launch_bounds__(256) void k_attn_score(
    const _Float16* __restrict__ encH, const float* __restrict__ q,
    float* __restrict__ scoreOut, int strideB, int L, int D)
{
  __shared__ float qs[1024];
  __shared__ float red[256];
  const int b = blockIdx.x, tid = threadIdx.x;
  for (int d = tid; d < D; d += 256) qs[d] = q[(size_t)b * D + d];
  __syncthreads();

  float raw[2];
#pragma unroll
  for (int it = 0; it < 2; ++it) {
    int l = tid + it * 256;
    const _Float16* e = encH + ((size_t)b * L + l) * D;
    float acc = 0.f;
    for (int d = 0; d < D; d += 8) {
      v8h ev = *(const v8h*)(e + d);
#pragma unroll
      for (int j = 0; j < 8; ++j) acc += (float)ev[j] * qs[d + j];
    }
    raw[it] = acc;
  }
  float mx = fmaxf(raw[0], raw[1]);
  red[tid] = mx; __syncthreads();
  for (int s = 128; s > 0; s >>= 1) {
    if (tid < s) red[tid] = fmaxf(red[tid], red[tid + s]);
    __syncthreads();
  }
  mx = red[0]; __syncthreads();
  float e0 = expf(raw[0] - mx), e1 = expf(raw[1] - mx);
  red[tid] = e0 + e1; __syncthreads();
  for (int s = 128; s > 0; s >>= 1) {
    if (tid < s) red[tid] += red[tid + s];
    __syncthreads();
  }
  float inv = 1.0f / red[0];
  scoreOut[(size_t)b * strideB + tid]       = e0 * inv;
  scoreOut[(size_t)b * strideB + tid + 256] = e1 * inv;
}

// weighted[b,d] = sum_l score[b,l] * enc[b,l,d]  -> xcat[b, 512+d] (fp16)
__global__ __launch_bounds__(256) void k_attn_weighted(
    const _Float16* __restrict__ encH, const float* __restrict__ score, int strideB,
    _Float16* __restrict__ xcat, int L, int D)
{
  __shared__ float ss[512];
  const int b = blockIdx.x, tid = threadIdx.x;
  for (int l = tid; l < L; l += 256) ss[l] = score[(size_t)b * strideB + l];
  __syncthreads();
  float acc[4] = {0.f, 0.f, 0.f, 0.f};
  const _Float16* e = encH + (size_t)b * L * D;
  for (int l = 0; l < L; ++l) {
    float s = ss[l];
    const _Float16* er = e + (size_t)l * D;
#pragma unroll
    for (int j = 0; j < 4; ++j) acc[j] += s * (float)er[tid + j * 256];
  }
#pragma unroll
  for (int j = 0; j < 4; ++j)
    xcat[(size_t)b * 2688 + 512 + tid + j * 256] = (_Float16)acc[j];
}

// ------------------------------- recurrence glue ---------------------------

__global__ void k_vis_copy(const _Float16* __restrict__ visH, _Float16* __restrict__ xcat,
                           int t, int T) {
  int b = blockIdx.x, tid = threadIdx.x;
  for (int j = tid; j < 512; j += 256)
    xcat[(size_t)b * 2688 + j] = visH[((size_t)b * T + t) * 512 + j];
}

__global__ void k_cont_copy(const _Float16* __restrict__ xcat, _Float16* __restrict__ cont) {
  int b = blockIdx.x, tid = threadIdx.x;
  for (int j = tid; j < 1664; j += 256)
    cont[(size_t)b * 2688 + 1024 + j] = xcat[(size_t)b * 2688 + j];
}

__global__ void k_lstm(const float* __restrict__ gates, float* __restrict__ hF,
                       float* __restrict__ cF, _Float16* __restrict__ xcat,
                       _Float16* __restrict__ cont) {
  int i = blockIdx.x * 256 + threadIdx.x;   // over B*1024
  int b = i >> 10, d = i & 1023;
  const float* g = gates + (size_t)b * 4096;
  float ig = 1.f / (1.f + expf(-g[d]));
  float fg = 1.f / (1.f + expf(-g[d + 1024]));
  float gg = tanhf(g[d + 2048]);
  float og = 1.f / (1.f + expf(-g[d + 3072]));
  float c = fg * cF[i] + ig * gg;
  float h = og * tanhf(c);
  cF[i] = c; hF[i] = h;
  xcat[(size_t)b * 2688 + 1664 + d] = (_Float16)h;  // h slot for next step
  cont[(size_t)b * 2688 + d]        = (_Float16)h;  // cont = [h_t, inp]
}

// argmax over V, then e_t = emb_w[argmax] -> xcat e-slot (fp16)
__global__ __launch_bounds__(256) void k_argmax(
    const float* __restrict__ act, int strideB, const float* __restrict__ embW,
    _Float16* __restrict__ xcat, int V, int E)
{
  __shared__ float rv[256];
  __shared__ int   ri[256];
  const int b = blockIdx.x, tid = threadIdx.x;
  float best = -3.0e38f; int bi = 0;
  for (int l = tid; l < V; l += 256) {
    float v = act[(size_t)b * strideB + l];
    if (v > best) { best = v; bi = l; }
  }
  rv[tid] = best; ri[tid] = bi; __syncthreads();
  for (int s = 128; s > 0; s >>= 1) {
    if (tid < s) {
      if (rv[tid + s] > rv[tid] || (rv[tid + s] == rv[tid] && ri[tid + s] < ri[tid])) {
        rv[tid] = rv[tid + s]; ri[tid] = ri[tid + s];
      }
    }
    __syncthreads();
  }
  int idx = ri[0];
  if (tid < E) xcat[(size_t)b * 2688 + 1536 + tid] = (_Float16)embW[(size_t)idx * E + tid];
}

// ------------------------------- launcher ----------------------------------

extern "C" void kernel_launch(void* const* d_in, const int* in_sizes, int n_in,
                              void* d_out, int out_size, void* d_ws, size_t ws_size,
                              hipStream_t stream) {
  (void)n_in; (void)out_size; (void)ws_size;

  const float* enc     = (const float*)d_in[0];
  const float* frames  = (const float*)d_in[1];
  const float* h0      = (const float*)d_in[2];
  const float* c0      = (const float*)d_in[3];
  /* d_in[4] = max_decode (== T in reference; reading it would break capture) */
  const float* emb_w   = (const float*)d_in[5];
  const float* go      = (const float*)d_in[6];
  const float* conv1_w = (const float*)d_in[7];
  const float* conv1_b = (const float*)d_in[8];
  const float* bn1_g   = (const float*)d_in[9];
  const float* bn1_b   = (const float*)d_in[10];
  const float* bn1_m   = (const float*)d_in[11];
  const float* bn1_v   = (const float*)d_in[12];
  const float* conv2_w = (const float*)d_in[13];
  const float* conv2_b = (const float*)d_in[14];
  const float* bn2_g   = (const float*)d_in[15];
  const float* bn2_b   = (const float*)d_in[16];
  const float* bn2_m   = (const float*)d_in[17];
  const float* bn2_v   = (const float*)d_in[18];
  const float* fc_w    = (const float*)d_in[19];
  const float* fc_b    = (const float*)d_in[20];
  const float* hfc_w   = (const float*)d_in[21];
  const float* hfc_b   = (const float*)d_in[22];
  const float* w_ih    = (const float*)d_in[23];
  const float* b_ih    = (const float*)d_in[24];
  const float* w_hh    = (const float*)d_in[25];
  const float* b_hh    = (const float*)d_in[26];
  const float* actor_w = (const float*)d_in[27];
  const float* actor_b = (const float*)d_in[28];

  const int B  = in_sizes[2] / 1024;          // 128
  const int L  = in_sizes[0] / (B * 1024);    // 512
  const int T  = in_sizes[1] / (B * 512 * 49);// 25
  const int V = 512, DH = 1024, DE = 128, DIN = 1664, DC = 2688;
  const int BT = B * T;                       // 3200
  const int M1 = BT * 49;                     // 156800

  // bump allocator over d_ws
  char* wsb = (char*)d_ws;
  size_t off = 0;
  auto alloc = [&](size_t bytes) -> void* {
    off = (off + 255) & ~(size_t)255;
    void* p = wsb + off;
    off += bytes;
    return p;
  };

  _Float16* encH = (_Float16*)alloc((size_t)B * L * DH * 2);
  _Float16* w1h  = (_Float16*)alloc((size_t)256 * 512 * 2);
  _Float16* w2h  = (_Float16*)alloc((size_t)64 * 256 * 2);
  _Float16* fch  = (_Float16*)alloc((size_t)512 * 3136 * 2);
  _Float16* hfch = (_Float16*)alloc((size_t)DH * DH * 2);
  _Float16* wcat = (_Float16*)alloc((size_t)4096 * DC * 2);
  _Float16* acth = (_Float16*)alloc((size_t)DE * DC * 2);
  _Float16* embh = (_Float16*)alloc((size_t)V * DE * 2);
  _Float16* a1   = (_Float16*)alloc((size_t)M1 * 256 * 2);
  _Float16* a2   = (_Float16*)alloc((size_t)BT * 3136 * 2);
  _Float16* visH = (_Float16*)alloc((size_t)BT * 512 * 2);
  _Float16* xcat = (_Float16*)alloc((size_t)B * DC * 2);  // [vis|weighted|e|h]
  _Float16* cont = (_Float16*)alloc((size_t)B * DC * 2);  // [h_t|inp]
  _Float16* aemb = (_Float16*)alloc((size_t)B * DE * 2);
  float* s1  = (float*)alloc(256 * 4);
  float* t1  = (float*)alloc(256 * 4);
  float* s2  = (float*)alloc(64 * 4);
  float* t2  = (float*)alloc(64 * 4);
  float* bg  = (float*)alloc(4096 * 4);
  float* q   = (float*)alloc((size_t)B * DH * 4);
  float* gts = (float*)alloc((size_t)B * 4096 * 4);
  float* hF  = (float*)alloc((size_t)B * DH * 4);
  float* cF  = (float*)alloc((size_t)B * DH * 4);

  float* actions = (float*)d_out;                       // [B,T,V]
  float* scores  = actions + (size_t)B * T * V;         // [B,T,L,1]
  float* outHC   = scores + (size_t)B * T * L;          // h then c

  auto cvt = [&](const float* s, _Float16* d, size_t n) {
    k_cvt_h<<<dim3((unsigned)((n + 255) / 256)), dim3(256), 0, stream>>>(s, d, (int)n);
  };

  // ---- prep: fp16 weights, folded BN, fused LSTM weight cat, init state
  cvt(conv1_w, w1h, (size_t)256 * 512);
  cvt(conv2_w, w2h, (size_t)64 * 256);
  cvt(fc_w,    fch, (size_t)512 * 3136);
  cvt(hfc_w,   hfch, (size_t)DH * DH);
  cvt(actor_w, acth, (size_t)DE * DC);
  cvt(emb_w,   embh, (size_t)V * DE);
  cvt(enc,     encH, (size_t)B * L * DH);
  k_wcat<<<dim3((4096 * DC + 255) / 256), dim3(256), 0, stream>>>(w_ih, w_hh, wcat, 4096, DIN, DH);
  k_bnprep<<<dim3(1), dim3(256), 0, stream>>>(bn1_g, bn1_b, bn1_m, bn1_v, conv1_b, s1, t1, 256);
  k_bnprep<<<dim3(1), dim3(256), 0, stream>>>(bn2_g, bn2_b, bn2_m, bn2_v, conv2_b, s2, t2, 64);
  k_addb<<<dim3(16), dim3(256), 0, stream>>>(b_ih, b_hh, bg, 4096);
  k_init<<<dim3((B * DH + 255) / 256), dim3(256), 0, stream>>>(h0, c0, go, xcat, hF, cF, B);

  // ---- phase 1: visual encoder for ALL frames (no recurrent dependency)
  // conv1: [156800,512] x [256,512]^T, BN+ReLU fused, fp16 out
  k_gemm<4><<<dim3(256 / 128, M1 / 128), dim3(256), 0, stream>>>(
      nullptr, 0, frames, w1h, s1, t1, nullptr, 0, a1, M1, 256, 512, 1, 1, 0);
  // conv2: [156800,256] x [64,256]^T, BN+ReLU, conv-flatten output [bt, c*49+hw]
  k_gemm<2><<<dim3(1, M1 / 128), dim3(256), 0, stream>>>(
      a1, 256, nullptr, w2h, s2, t2, nullptr, 0, a2, M1, 64, 256, 1, 0, 1);
  // fc: [3200,3136] x [512,3136]^T + b -> visH fp16
  k_gemm<4><<<dim3(512 / 128, BT / 128), dim3(256), 0, stream>>>(
      a2, 3136, nullptr, fch, nullptr, fc_b, nullptr, 0, visH, BT, 512, 3136, 0, 0, 0);

  // ---- phase 2: sequential decode
  for (int t = 0; t < T; ++t) {
    k_vis_copy<<<dim3(B), dim3(256), 0, stream>>>(visH, xcat, t, T);
    // q = h @ hfc_w^T + b   (A = h slot inside xcat, lda = 2688)
    k_gemm<4><<<dim3(DH / 128, B / 128), dim3(256), 0, stream>>>(
        xcat + 1664, DC, nullptr, hfch, nullptr, hfc_b, q, DH, nullptr, B, DH, DH, 0, 0, 0);
    k_attn_score<<<dim3(B), dim3(256), 0, stream>>>(
        encH, q, scores + (size_t)t * L, T * L, L, DH);
    k_attn_weighted<<<dim3(B), dim3(256), 0, stream>>>(
        encH, scores + (size_t)t * L, T * L, xcat, L, DH);
    k_cont_copy<<<dim3(B), dim3(256), 0, stream>>>(xcat, cont);
    // gates = [inp|h] @ [w_ih|w_hh]^T + (b_ih+b_hh)
    k_gemm<4><<<dim3(4096 / 128, B / 128), dim3(256), 0, stream>>>(
        xcat, DC, nullptr, wcat, nullptr, bg, gts, 4096, nullptr, B, 4096, DC, 0, 0, 0);
    k_lstm<<<dim3((B * DH) / 256), dim3(256), 0, stream>>>(gts, hF, cF, xcat, cont);
    // action_emb = [h_t|inp] @ actor_w^T + b
    k_gemm<4><<<dim3(DE / 128, B / 128), dim3(256), 0, stream>>>(
        cont, DC, nullptr, acth, nullptr, actor_b, nullptr, 0, aemb, B, DE, DC, 0, 0, 0);
    // action = action_emb @ emb_w^T -> actions[b,t,:]
    k_gemm<4><<<dim3(V / 128, B / 128), dim3(256), 0, stream>>>(
        aemb, DE, nullptr, embh, nullptr, nullptr, actions + (size_t)t * V, T * V,
        nullptr, B, V, DE, 0, 0, 0);
    k_argmax<<<dim3(B), dim3(256), 0, stream>>>(
        actions + (size_t)t * V, T * V, emb_w, xcat, V, DE);
  }

  // ---- final state outputs
  k_copy<<<dim3((B * DH + 255) / 256), dim3(256), 0, stream>>>(hF, outHC, B * DH);
  k_copy<<<dim3((B * DH + 255) / 256), dim3(256), 0, stream>>>(cF, outHC + (size_t)B * DH, B * DH);
}
